// ScaledDotProductAttention_927712936610
// MI455X (gfx1250) — compile-verified, matrix operand order checked by reading, not run on hardware
//
#include <hip/hip_runtime.h>
#include <hip/hip_bf16.h>

typedef __attribute__((ext_vector_type(16))) _Float16 v16h;
typedef __attribute__((ext_vector_type(8)))  float    v8f;

#define B_SZ  128
#define S_SZ  1024
#define D_SZ  128
#define MTILE 64
#define NTILE 64
#define NWAVES 4
#define QK_SCALE 0.08838834764831845f   /* 1/sqrt(128) */

// Manual LDS layout (single 64KB block):
//   [0,16K)   sK  : K chunk,  f16 row-major [64 keys][128 d]
//   [16K,32K) sVt : V chunk,  f16 transposed [128 d][64 keys]
//   [32K,48K) per-wave 4KB: scores f32 [16][64], OVERLAID by probs f16 [16][64]
//             (safe: every score load is a data dependency of m_new, which all
//              P stores depend on, so all DS reads precede all DS writes)
//   [48K,64K) per-wave 4KB: dropout mask f32 [16][64], filled by async copies
#define OFF_K   0
#define OFF_VT  16384
#define OFF_SP  32768
#define OFF_M   49152
#define SMEM_BYTES 65536

// Load a 16x32 f16 fragment (A layout, or B layout of the transpose) from LDS.
// Per ISA 7.12.2 (16-bit A-matrix 16x32): lane holds row (lane&15); VGPR j,
// half p -> k = ((j>>2)<<4) | (hi<<3) | ((j&3)<<1) | p. Half-pairs are
// contiguous in k, so each dword is one aligned b32 LDS load (compiler merges
// adjacent ones into ds_load_b128).
__device__ __forceinline__ v16h lds_frag(const _Float16* rowp, int c0, int hi) {
  union { v16h h; unsigned u[8]; } t;
#pragma unroll
  for (int j = 0; j < 8; ++j) {
    int k = ((j >> 2) << 4) | (hi << 3) | ((j & 3) << 1);
    t.u[j] = *(const unsigned*)(rowp + c0 + k);
  }
  return t.h;
}

__global__ __launch_bounds__(128)
void fa_fwd_kernel(const float* __restrict__ Q, const float* __restrict__ K,
                   const float* __restrict__ V, const float* __restrict__ Mask,
                   float* __restrict__ Out) {
  __shared__ char smem[SMEM_BYTES] __attribute__((aligned(16)));

  const int tid  = threadIdx.x;
  const int wave = tid >> 5;
  const int lane = tid & 31;
  const int lo   = lane & 15;
  const int hi   = lane >> 4;

  const int b     = blockIdx.y;
  const int qrow0 = blockIdx.x * MTILE + wave * 16;   // this wave's 16 query rows

  _Float16* sK  = (_Float16*)(smem + OFF_K);
  _Float16* sVt = (_Float16*)(smem + OFF_VT);
  float*    sSw = (float*)(smem + OFF_SP + wave * 4096);     // per-wave scores
  _Float16* sPw = (_Float16*)(smem + OFF_SP + wave * 4096);  // overlay: probs f16
  float*    sMw = (float*)(smem + OFF_M + wave * 4096);      // per-wave mask tile

  // LDS byte address for async-copy destination: flat LDS-aperture pointers map
  // to LDS by their low 32 bits (ISA 10.2 aperture rules).
  const unsigned mlds0 =
      (unsigned)(size_t)sMw + (unsigned)((hi * 64 + lo * 4) * 4);
  const float* gmask0 =
      Mask + ((size_t)b * S_SZ + qrow0 + hi) * (size_t)S_SZ + lo * 4;

  // ---- Q fragments: load f32, fold in 1/sqrt(D), convert to f16, keep in regs ----
  v16h qf[4];
  {
    const float* qp = Q + ((size_t)b * S_SZ + qrow0 + lo) * D_SZ;
#pragma unroll
    for (int ks = 0; ks < 4; ++ks) {
      union { v16h h; _Float16 e[16]; } t;
#pragma unroll
      for (int j = 0; j < 8; ++j) {
        int k = (ks << 5) + (((j >> 2) << 4) | (hi << 3) | ((j & 3) << 1));
        t.e[2 * j]     = (_Float16)(qp[k]     * QK_SCALE);
        t.e[2 * j + 1] = (_Float16)(qp[k + 1] * QK_SCALE);
      }
      qf[ks] = t.h;
    }
  }

  v8f zero = {};
  v8f o[8];
#pragma unroll
  for (int d = 0; d < 8; ++d) o[d] = zero;
  float m_run = -__builtin_inff();
  float l_run = 0.0f;

  for (int kb = 0; kb < S_SZ; kb += NTILE) {
    // ---- async-copy this wave's 16x64 f32 mask tile into LDS (ASYNCcnt).
    // Issued before the staging barriers + QK^T WMMAs so the 512MB mask stream
    // overlaps compute. Each b128 op moves a contiguous 512B slab (2 rows).
#pragma unroll
    for (int i = 0; i < 8; ++i) {
      const float* ga = gmask0 + (size_t)(2 * i) * S_SZ + kb;
      unsigned ldso = mlds0 + (unsigned)(i * 512);
      asm volatile("global_load_async_to_lds_b128 %0, %1, off"
                   :: "v"(ldso), "v"(ga) : "memory");
    }

    __syncthreads();                       // WAR protection on sK/sVt
    // ---- stage K (row-major) and V (transposed) as f16 ----
    {
      const float* kp = K + ((size_t)b * S_SZ + kb) * D_SZ;
      const float* vp = V + ((size_t)b * S_SZ + kb) * D_SZ;
      for (int i = tid; i < NTILE * D_SZ; i += 128) {
        int r = i >> 7, c = i & (D_SZ - 1);
        sK[i] = (_Float16)kp[i];
        sVt[c * NTILE + r] = (_Float16)vp[i];
      }
    }
    __syncthreads();

    // ---- scores: S(16x64) = Q(16x128) * K^T, f32 accum via WMMA ----
    v8f sacc[4];
#pragma unroll
    for (int ns = 0; ns < 4; ++ns) {
      v8f acc = zero;
#pragma unroll
      for (int ks = 0; ks < 4; ++ks) {
        v16h bk = lds_frag(sK + (ns * 16 + lo) * D_SZ, ks * 32, hi);
        acc = __builtin_amdgcn_wmma_f32_16x16x32_f16(false, qf[ks], false, bk,
                                                     (short)0, acc, false, false);
      }
      sacc[ns] = acc;
    }
    // C layout (lane = col, VGPR j = row j+8*hi) -> row-major LDS scratch
#pragma unroll
    for (int ns = 0; ns < 4; ++ns)
#pragma unroll
      for (int j = 0; j < 8; ++j)
        sSw[(j + 8 * hi) * NTILE + ns * 16 + lo] = sacc[ns][j];
    __syncthreads();

    // ---- online softmax (+ dropout mask applied AFTER exp, excluded from sum) ----
    float alpha;
    {
      const int r  = lo;           // lanes r and r+16 share row r (32-col halves)
      const int c0 = hi * 32;
      const float* sw = sSw + r * NTILE + c0;
      float4 sv[8];
      float mx = -__builtin_inff();
#pragma unroll
      for (int i = 0; i < 8; ++i) {
        sv[i] = ((const float4*)sw)[i];
        mx = fmaxf(mx, fmaxf(fmaxf(sv[i].x, sv[i].y), fmaxf(sv[i].z, sv[i].w)));
      }
      mx = fmaxf(mx, __shfl_xor(mx, 16, 32));
      float m_new = fmaxf(m_run, mx);
      alpha = __expf(m_run - m_new);

      asm volatile("s_wait_asynccnt 0x0" ::: "memory");  // mask tile landed
      const float4* mrow = (const float4*)(sMw + r * NTILE + c0);

      float lsum = 0.0f;
      unsigned* pw = (unsigned*)(sPw + r * NTILE + c0);
#pragma unroll
      for (int i = 0; i < 8; ++i) {
        float4 p;
        p.x = __expf(sv[i].x - m_new); p.y = __expf(sv[i].y - m_new);
        p.z = __expf(sv[i].z - m_new); p.w = __expf(sv[i].w - m_new);
        lsum += p.x + p.y + p.z + p.w;       // denominator excludes dropout mask
        float4 mk = mrow[i];
        p.x *= mk.x; p.y *= mk.y; p.z *= mk.z; p.w *= mk.w;
        _Float16 h0 = (_Float16)p.x, h1 = (_Float16)p.y;
        _Float16 h2 = (_Float16)p.z, h3 = (_Float16)p.w;
        pw[2 * i] = (unsigned)__builtin_bit_cast(unsigned short, h0) |
                    ((unsigned)__builtin_bit_cast(unsigned short, h1) << 16);
        pw[2 * i + 1] = (unsigned)__builtin_bit_cast(unsigned short, h2) |
                        ((unsigned)__builtin_bit_cast(unsigned short, h3) << 16);
      }
      lsum += __shfl_xor(lsum, 16, 32);
      l_run = l_run * alpha + lsum;
      m_run = m_new;
    }
    __syncthreads();

    // ---- rescale O by alpha (per-row broadcast) and accumulate O += P*V ----
    {
#pragma unroll
      for (int j = 0; j < 8; ++j) {
        float aj = __shfl(alpha, j + 8 * hi, 32);
#pragma unroll
        for (int d = 0; d < 8; ++d) o[d][j] *= aj;
      }
      v16h pf[2];
#pragma unroll
      for (int ks = 0; ks < 2; ++ks)
        pf[ks] = lds_frag(sPw + lo * NTILE, ks * 32, hi);
#pragma unroll
      for (int d = 0; d < 8; ++d) {
#pragma unroll
        for (int ks = 0; ks < 2; ++ks) {
          v16h bv = lds_frag(sVt + (d * 16 + lo) * NTILE, ks * 32, hi);
          o[d] = __builtin_amdgcn_wmma_f32_16x16x32_f16(false, pf[ks], false, bv,
                                                        (short)0, o[d], false, false);
        }
      }
    }
  }

  // ---- normalize by softmax denominator and store ----
  float inv = 1.0f / l_run;   // lane r (and r+16) holds row r's denominator
  float invj[8];
#pragma unroll
  for (int j = 0; j < 8; ++j) invj[j] = __shfl(inv, j + 8 * hi, 32);
  float* op = Out + ((size_t)b * S_SZ + qrow0) * D_SZ;
#pragma unroll
  for (int d = 0; d < 8; ++d)
#pragma unroll
    for (int j = 0; j < 8; ++j)
      op[(j + 8 * hi) * D_SZ + d * 16 + lo] = o[d][j] * invj[j];
}

extern "C" void kernel_launch(void* const* d_in, const int* in_sizes, int n_in,
                              void* d_out, int out_size, void* d_ws, size_t ws_size,
                              hipStream_t stream) {
  const float* Q = (const float*)d_in[0];
  const float* K = (const float*)d_in[1];
  const float* V = (const float*)d_in[2];
  const float* M = (const float*)d_in[3];
  float* O = (float*)d_out;
  dim3 grid(S_SZ / MTILE, B_SZ);
  fa_fwd_kernel<<<grid, dim3(128), 0, stream>>>(Q, K, V, M, O);
}
